// CharRNN_69097433858414
// MI455X (gfx1250) — matrix operand order repaired and checked by reading, not compiled, not run on hardware
//
#include <hip/hip_runtime.h>
#include <hip/hip_bf16.h>
#include <math.h>

#define BB 64
#define TT 128
#define HH 512
#define VV 8000
#define NTOK (BB*TT)     // 8192
#define G4 (4*HH)        // 2048
#define NT_V (VV/16)     // 500 vocab tiles
#define LDA 520          // padded LDS row stride (elems): 1040B -> +4 banks/row

typedef __bf16 bf_t;
typedef __attribute__((ext_vector_type(16))) __bf16 v16bf;
typedef __attribute__((ext_vector_type(8)))  float  v8f;

__device__ __forceinline__ v8f vzero(){
  v8f z = {0.f,0.f,0.f,0.f,0.f,0.f,0.f,0.f};
  return z;
}

__device__ __forceinline__ float sigm(float x){ return 1.f/(1.f+__expf(-x)); }

// ---- WMMA helpers --------------------------------------------------------
// B fragment: pre-packed so lane L reads 16 contiguous bf16 (32B):
//   n = n_tile*16 + (L&15), k = k_tile*32 + (L>=16?16:0) + j, j=0..15
__device__ __forceinline__ v16bf load_b_frag(const bf_t* P, int tile){
  const bf_t* q = P + ((size_t)tile << 9) + ((threadIdx.x & 31) << 4);
  union { uint4 a[2]; v16bf v; } u;
  u.a[0] = *(const uint4*)q;
  u.a[1] = *(const uint4*)(q + 8);
  return u.v;
}

// A fragment from bf16 row-major [M x lda]: lane L covers row m0+(L&15),
// K {k0+h*16..+15}, h = L>>4  (two 16B loads)
__device__ __forceinline__ v16bf load_a_frag(const bf_t* A, int lda, int m0, int k0){
  int lane = threadIdx.x & 31;
  const bf_t* base = A + (size_t)(m0 + (lane & 15))*lda + k0 + ((lane >> 4) << 4);
  union { uint4 a[2]; v16bf v; } u;
  u.a[0] = *(const uint4*)base;
  u.a[1] = *(const uint4*)(base + 8);
  return u.v;
}

__device__ __forceinline__ v8f wmma_bf16(v16bf a, v16bf b, v8f c){
  return __builtin_amdgcn_wmma_f32_16x16x32_bf16(false, a, false, b, (short)0, c, false, false);
}

// ---- async stage: rows x 1024B from global into padded LDS tile ---------
__device__ __forceinline__ void stage_async(const bf_t* src, bf_t* dstLds, int rows){
  const int perRow = 1024 >> 4;              // 64 b128 transfers per row
  int total = rows * perRow;
  for(int li = threadIdx.x; li < total; li += blockDim.x){
    int row = li / perRow;
    int off = (li % perRow) << 4;
    unsigned long long g = (unsigned long long)(uintptr_t)src
                         + (unsigned long long)row*1024ull + (unsigned)off;
    unsigned int l = (unsigned int)(uintptr_t)(dstLds) + (unsigned)(row*(LDA*2) + off);
    asm volatile("global_load_async_to_lds_b128 %0, %1, off"
                 :: "v"(l), "v"(g) : "memory");
  }
  asm volatile("s_wait_asynccnt 0" ::: "memory");
  __syncthreads();
}

// ---- init / pack / embed -------------------------------------------------
__global__ void __launch_bounds__(256) k_init(float* c1, float* c2, bf_t* h1, bf_t* h2, float* out){
  int i = blockIdx.x*blockDim.x + threadIdx.x;
  if(i < BB*HH){ c1[i]=0.f; c2[i]=0.f; h1[i]=(bf_t)0.f; h2[i]=(bf_t)0.f; }
  if(i==0) out[0]=0.f;
}

__global__ void __launch_bounds__(256) k_pack(const float* W, bf_t* out, int NS, int k_off, long total){
  long o = (long)blockIdx.x*blockDim.x + threadIdx.x;
  if(o >= total) return;
  int  j    = (int)(o & 15);
  int  lane = (int)((o >> 4) & 31);
  long rest = o >> 9;
  int  kt   = (int)(rest & 15);        // K/32 == 16 (K = 512)
  long nt   = rest >> 4;
  int  k    = k_off + kt*32 + ((lane >> 4) << 4) + j;
  long n    = nt*16 + (lane & 15);
  out[o] = (bf_t)W[(long)k*NS + n];
}

__global__ void __launch_bounds__(256) k_embed(const int* ids, const float* emb, bf_t* xbf){
  int r = blockIdx.x;
  int t = r >> 6, b = r & 63;
  int tok = ids[b*TT + t];
  const float* src = emb + (long)tok*HH;
  bf_t* dst = xbf + (long)r*HH;
  for(int c = threadIdx.x; c < HH; c += blockDim.x) dst[c] = (bf_t)src[c];
}

// ---- X1 = x @ Wx1 + b1  (8192 x 512 x 2048), block tile 64x128 -----------
__global__ void __launch_bounds__(256) k_gemm_x1(const bf_t* xbf, const bf_t* wp,
                                                 const float* b1, float* X1){
  extern __shared__ char smem[];
  bf_t* As = (bf_t*)smem;                           // 64 x LDA
  stage_async(xbf + (size_t)blockIdx.y*64*HH, As, 64);

  int w    = threadIdx.x >> 5;
  int lane = threadIdx.x & 31;
  int m0l  = (w & 3)*16;
  int ntb  = blockIdx.x*8 + (w >> 2)*4;
  v8f acc[4];
  #pragma unroll
  for(int i=0;i<4;i++) acc[i] = vzero();

  // unroll-by-2 software pipeline, alternating buffers (no register copies)
  v16bf ba[4], bb[4];
  #pragma unroll
  for(int i=0;i<4;i++) ba[i] = load_b_frag(wp, (ntb+i)*16);
  for(int kt=0; kt<16; kt+=2){
    #pragma unroll
    for(int i=0;i<4;i++) bb[i] = load_b_frag(wp, (ntb+i)*16 + kt + 1);
    v16bf a0 = load_a_frag(As, LDA, m0l, kt*32);
    #pragma unroll
    for(int i=0;i<4;i++) acc[i] = wmma_bf16(a0, ba[i], acc[i]);
    if(kt + 2 < 16){
      #pragma unroll
      for(int i=0;i<4;i++) ba[i] = load_b_frag(wp, (ntb+i)*16 + kt + 2);
    }
    v16bf a1 = load_a_frag(As, LDA, m0l, (kt+1)*32);
    #pragma unroll
    for(int i=0;i<4;i++) acc[i] = wmma_bf16(a1, bb[i], acc[i]);
  }
  int hi = lane >> 4, nl = lane & 15;
  int mg = blockIdx.y*64 + m0l + hi*8;
  #pragma unroll
  for(int i=0;i<4;i++){
    int n = (ntb+i)*16 + nl;
    float bias = b1[n];
    #pragma unroll
    for(int r=0;r<8;r++)
      X1[(long)(mg + r)*G4 + n] = acc[i][r] + bias;
  }
}

// K-loop body shared by the recurrent kernels: acc += A(LDS) @ Wp[nt]
__device__ __forceinline__ void gemm_k512(v8f* acc, const bf_t* As,
                                          const bf_t* wp, int nt){
  v16bf b0 = load_b_frag(wp, nt*16);
  for(int kt=0; kt<16; kt+=2){
    v16bf b1 = load_b_frag(wp, nt*16 + kt + 1);
    #pragma unroll
    for(int m=0;m<4;m++){
      v16bf a = load_a_frag(As, LDA, m*16, kt*32);
      acc[m] = wmma_bf16(a, b0, acc[m]);
    }
    if(kt + 2 < 16) b0 = load_b_frag(wp, nt*16 + kt + 2);
    #pragma unroll
    for(int m=0;m<4;m++){
      v16bf a = load_a_frag(As, LDA, m*16, (kt+1)*32);
      acc[m] = wmma_bf16(a, b1, acc[m]);
    }
  }
}

// ---- LSTM step layer 1: gates = X1[t] + h_prev @ Wh1; pointwise ---------
__global__ void __launch_bounds__(256) k_step1(int t, const bf_t* hprev, bf_t* hnew,
                                               float* c1, const float* X1,
                                               const float* b1, const bf_t* whp){
  extern __shared__ char smem[];
  bf_t*  As   = (bf_t*)smem;                        // 64 x LDA (66560 B)
  float* gbuf = (float*)(smem + 64*LDA*2);          // [4][64][32]
  stage_async(hprev, As, 64);

  int w = threadIdx.x >> 5, lane = threadIdx.x & 31;
  int hc0 = blockIdx.x*32;
  int g   = w >> 1;
  int ntl = w & 1;
  int nt  = (g*HH + hc0 + ntl*16) >> 4;
  v8f acc[4];
  #pragma unroll
  for(int m=0;m<4;m++) acc[m] = vzero();
  gemm_k512(acc, As, whp, nt);

  int hi = lane >> 4, nl = lane & 15;
  int n  = g*HH + hc0 + ntl*16 + nl;
  float bias = b1[n];
  #pragma unroll
  for(int m=0;m<4;m++){
    int mb = m*16 + hi*8;
    #pragma unroll
    for(int r=0;r<8;r++){
      int row = mb + r;
      gbuf[(g*64 + row)*32 + ntl*16 + nl] =
          acc[m][r] + bias + X1[(long)(t*64 + row)*G4 + n];
    }
  }
  __syncthreads();
  for(int idx = threadIdx.x; idx < 64*32; idx += blockDim.x){
    int row = idx >> 5, c = idx & 31;
    float gi = gbuf[(0*64+row)*32+c], gj = gbuf[(1*64+row)*32+c];
    float gf = gbuf[(2*64+row)*32+c], go = gbuf[(3*64+row)*32+c];
    long ci = (long)row*HH + hc0 + c;
    float cn = c1[ci]*sigm(gf) + sigm(gi)*tanhf(gj);
    c1[ci] = cn;
    hnew[ci] = (bf_t)(tanhf(cn)*sigm(go));
  }
}

// ---- LSTM step layer 2: gates = h1 @ Wx2 + h2_prev @ Wh2; pointwise -----
__global__ void __launch_bounds__(256) k_step2(int t, const bf_t* h1, const bf_t* h2prev,
                                               bf_t* h2new, float* c2, const float* b2,
                                               const bf_t* wxp, const bf_t* whp, bf_t* hsbf){
  extern __shared__ char smem[];
  bf_t*  As   = (bf_t*)smem;
  float* gbuf = (float*)(smem + 64*LDA*2);
  int w = threadIdx.x >> 5, lane = threadIdx.x & 31;
  int hc0 = blockIdx.x*32;
  int g   = w >> 1;
  int ntl = w & 1;
  int nt  = (g*HH + hc0 + ntl*16) >> 4;
  v8f acc[4];
  #pragma unroll
  for(int m=0;m<4;m++) acc[m] = vzero();

  stage_async(h1, As, 64);                // phase 1: h1 @ Wx2
  gemm_k512(acc, As, wxp, nt);
  __syncthreads();                        // all waves done reading As
  stage_async(h2prev, As, 64);            // phase 2: h2_prev @ Wh2
  gemm_k512(acc, As, whp, nt);

  int hi = lane >> 4, nl = lane & 15;
  int n  = g*HH + hc0 + ntl*16 + nl;
  float bias = b2[n];
  #pragma unroll
  for(int m=0;m<4;m++){
    int mb = m*16 + hi*8;
    #pragma unroll
    for(int r=0;r<8;r++)
      gbuf[(g*64 + mb + r)*32 + ntl*16 + nl] = acc[m][r] + bias;
  }
  __syncthreads();
  for(int idx = threadIdx.x; idx < 64*32; idx += blockDim.x){
    int row = idx >> 5, c = idx & 31;
    float gi = gbuf[(0*64+row)*32+c], gj = gbuf[(1*64+row)*32+c];
    float gf = gbuf[(2*64+row)*32+c], go = gbuf[(3*64+row)*32+c];
    long ci = (long)row*HH + hc0 + c;
    float cn = c2[ci]*sigm(gf) + sigm(gi)*tanhf(gj);
    c2[ci] = cn;
    float hn = tanhf(cn)*sigm(go);
    h2new[ci] = (bf_t)hn;
    hsbf[(long)(t*64 + row)*HH + hc0 + c] = (bf_t)hn;
  }
}

// ---- Fused logits GEMM + online log-softmax cross-entropy ---------------
__global__ void __launch_bounds__(256) k_ce(const bf_t* hsbf, const bf_t* swp,
                                            const float* sb, const int* targets,
                                            float* out){
  extern __shared__ char smem[];
  bf_t*  As = (bf_t*)smem;                          // 64 x LDA
  float* Mw = (float*)(smem + 64*LDA*2);            // [8][64]
  float* Sw = Mw + 8*64;
  float* tg = Sw + 8*64;
  float* red = tg + 64;
  int*   lb  = (int*)(red + 256);                   // per-row labels

  int w = threadIdx.x >> 5, lane = threadIdx.x & 31;
  int r0 = blockIdx.x * 64;
  for(int i = threadIdx.x; i < 8*64; i += blockDim.x){ Mw[i] = -INFINITY; Sw[i] = 0.f; }
  if(threadIdx.x < 64){
    tg[threadIdx.x] = 0.f;
    int R = r0 + threadIdx.x;
    lb[threadIdx.x] = targets[(R & 63)*TT + (R >> 6)];
  }
  __syncthreads();
  stage_async(hsbf + (size_t)r0*HH, As, 64);

  int hi = lane >> 4, nl = lane & 15;
  for(int it = 0; it < (NT_V + 7)/8; ++it){
    int nt = it*8 + w;                  // wave-uniform
    if(nt < NT_V){
      if(nt + 8 < NT_V){
        const char* pf = (const char*)(swp + (((size_t)(nt+8)*16) << 9));
        __builtin_prefetch(pf + (threadIdx.x & 31)*512, 0, 3);
      }
      v8f acc[4];
      #pragma unroll
      for(int m=0;m<4;m++) acc[m] = vzero();
      gemm_k512(acc, As, swp, nt);

      int n = nt*16 + nl;
      float bias = sb[n];
      #pragma unroll
      for(int m=0;m<4;m++){
        #pragma unroll
        for(int r=0;r<8;r++){
          float v = acc[m][r] + bias;
          int row = m*16 + hi*8 + r;
          if(lb[row] == n) tg[row] = v;
          float mx = v;
          mx = fmaxf(mx, __shfl_xor(mx, 1, 32));
          mx = fmaxf(mx, __shfl_xor(mx, 2, 32));
          mx = fmaxf(mx, __shfl_xor(mx, 4, 32));
          mx = fmaxf(mx, __shfl_xor(mx, 8, 32));
          float p = __expf(v - mx);
          p += __shfl_xor(p, 1, 32);
          p += __shfl_xor(p, 2, 32);
          p += __shfl_xor(p, 4, 32);
          p += __shfl_xor(p, 8, 32);
          if(nl == 0){
            float M = Mw[w*64+row], S = Sw[w*64+row];
            float nM = fmaxf(M, mx);
            Sw[w*64+row] = S*__expf(M - nM) + p*__expf(mx - nM);
            Mw[w*64+row] = nM;
          }
        }
      }
    }
  }
  __syncthreads();
  float part = 0.f;
  if(threadIdx.x < 64){
    int row = threadIdx.x;
    float M = -INFINITY;
    #pragma unroll
    for(int ww=0; ww<8; ++ww) M = fmaxf(M, Mw[ww*64+row]);
    float S = 0.f;
    #pragma unroll
    for(int ww=0; ww<8; ++ww) S += Sw[ww*64+row]*__expf(Mw[ww*64+row] - M);
    part = (M + __logf(S)) - tg[row];
  }
  red[threadIdx.x] = part;
  __syncthreads();
  for(int s = 128; s > 0; s >>= 1){
    if(threadIdx.x < s) red[threadIdx.x] += red[threadIdx.x + s];
    __syncthreads();
  }
  if(threadIdx.x == 0) atomicAdd(out, red[0] / (float)NTOK);
}

// ---- host-side launcher --------------------------------------------------
extern "C" void kernel_launch(void* const* d_in, const int* in_sizes, int n_in,
                              void* d_out, int out_size, void* d_ws, size_t ws_size,
                              hipStream_t stream){
  const int*   ids = (const int*)d_in[0];
  const int*   tgt = (const int*)d_in[1];
  const float* emb = (const float*)d_in[2];
  const float* W1  = (const float*)d_in[3];
  const float* b1  = (const float*)d_in[4];
  const float* W2  = (const float*)d_in[5];
  const float* b2  = (const float*)d_in[6];
  const float* SW  = (const float*)d_in[7];
  const float* sb  = (const float*)d_in[8];
  float* out = (float*)d_out;

  char* p = (char*)d_ws;
  auto alloc = [&](size_t bytes)->char*{
    char* q = p; p += (bytes + 255) & ~(size_t)255; return q;
  };
  const size_t WSZ = (size_t)HH*G4*sizeof(bf_t);
  bf_t* Wx1p = (bf_t*)alloc(WSZ);
  bf_t* Wh1p = (bf_t*)alloc(WSZ);
  bf_t* Wx2p = (bf_t*)alloc(WSZ);
  bf_t* Wh2p = (bf_t*)alloc(WSZ);
  bf_t* SWp  = (bf_t*)alloc((size_t)HH*VV*sizeof(bf_t));
  bf_t* Xbf  = (bf_t*)alloc((size_t)NTOK*HH*sizeof(bf_t));
  float* X1  = (float*)alloc((size_t)NTOK*G4*sizeof(float));
  bf_t* hsbf = (bf_t*)alloc((size_t)NTOK*HH*sizeof(bf_t));
  bf_t* h1b[2] = { (bf_t*)alloc((size_t)BB*HH*sizeof(bf_t)),
                   (bf_t*)alloc((size_t)BB*HH*sizeof(bf_t)) };
  bf_t* h2b[2] = { (bf_t*)alloc((size_t)BB*HH*sizeof(bf_t)),
                   (bf_t*)alloc((size_t)BB*HH*sizeof(bf_t)) };
  float* c1 = (float*)alloc((size_t)BB*HH*sizeof(float));
  float* c2 = (float*)alloc((size_t)BB*HH*sizeof(float));

  const size_t ASZ   = (size_t)64*LDA*2;            // 66560 B staged A tile
  const size_t SH_X1 = ASZ;
  const size_t SH_ST = ASZ + (size_t)4*64*32*4;     // + gate buffer
  const size_t SH_CE = ASZ + (8*64 + 8*64 + 64 + 256 + 64)*sizeof(float);

  k_init<<<128, 256, 0, stream>>>(c1, c2, h1b[0], h2b[0], out);

  const long WTOT = (long)HH*G4;
  const long STOT = (long)HH*VV;
  k_pack<<<(int)((WTOT+255)/256), 256, 0, stream>>>(W1, Wx1p, G4, 0,  WTOT);
  k_pack<<<(int)((WTOT+255)/256), 256, 0, stream>>>(W1, Wh1p, G4, HH, WTOT);
  k_pack<<<(int)((WTOT+255)/256), 256, 0, stream>>>(W2, Wx2p, G4, 0,  WTOT);
  k_pack<<<(int)((WTOT+255)/256), 256, 0, stream>>>(W2, Wh2p, G4, HH, WTOT);
  k_pack<<<(int)((STOT+255)/256), 256, 0, stream>>>(SW, SWp,  VV, 0,  STOT);

  k_embed<<<NTOK, 256, 0, stream>>>(ids, emb, Xbf);

  k_gemm_x1<<<dim3(G4/128, NTOK/64), 256, SH_X1, stream>>>(Xbf, Wx1p, b1, X1);

  for(int t = 0; t < TT; ++t){
    int pp = t & 1, nn = pp ^ 1;
    k_step1<<<16, 256, SH_ST, stream>>>(t, h1b[pp], h1b[nn], c1, X1, b1, Wh1p);
    k_step2<<<16, 256, SH_ST, stream>>>(t, h1b[nn], h2b[pp], h2b[nn], c2, b2,
                                        Wx2p, Wh2p, hsbf);
  }

  k_ce<<<NTOK/64, 256, SH_CE, stream>>>(hsbf, SWp, sb, tgt, out);

  (void)in_sizes; (void)n_in; (void)out_size; (void)ws_size;
}
